// NodeTaskHead_3874060501244
// MI455X (gfx1250) — matrix operand order repaired and checked by reading, not compiled, not run on hardware
//
#include <hip/hip_runtime.h>
#include <hip/hip_bf16.h>

// ---------------------------------------------------------------------------
// NodeTaskHead force head, MI455X (gfx1250, wave32, WMMA).
//   force[b,n,c] = sum_{h,m} softmax(qk^T + bias)[b,h,n,m]
//                  * delta_pos[b,n,m,c] * (v[b,h,m,:] . Wf[c, h-block]) + bf[c]
// HBM-bound on attn_bias (134MB); bf16 WMMA for both GEMMs; flash-style
// online softmax fused with the 3-axis contraction (never materializes probs).
// k head-slices staged to LDS with GLOBAL_LOAD_ASYNC_TO_LDS_B128 (ASYNCcnt).
// ---------------------------------------------------------------------------

typedef __attribute__((ext_vector_type(16))) __bf16 v16bf;
typedef __attribute__((ext_vector_type(8)))  float v8f;

#define Bc 4
#define Nn 512
#define Ee 768
#define Hh 32
#define Dd 24
#define DP 32          // padded head dim for fragment-friendly loads

__device__ __forceinline__ unsigned short f32_to_bf16_bits(float x) {
    union { float f; unsigned u; } cv; cv.f = x;
    unsigned u = cv.u + 0x7fffu + ((cv.u >> 16) & 1u);   // RNE
    return (unsigned short)(u >> 16);
}
__device__ __forceinline__ __bf16 bits_to_bf16(unsigned short s) {
    union { unsigned short u; __bf16 b; } cv; cv.u = s; return cv.b;
}

// ---------------------------------------------------------------------------
// Kernel 0: zero the padded bf16 q/k buffers, init out with bf broadcast.
// ---------------------------------------------------------------------------
__global__ void init_kernel(unsigned* __restrict__ qk_zero, size_t nwords,
                            float* __restrict__ out, const float* __restrict__ bf) {
    size_t i = (size_t)blockIdx.x * blockDim.x + threadIdx.x;
    size_t stride = (size_t)gridDim.x * blockDim.x;
    for (size_t j = i; j < nwords; j += stride) qk_zero[j] = 0u;
    const size_t nout = (size_t)Bc * Nn * 3;
    for (size_t j = i; j < nout; j += stride) out[j] = bf[j % 3];
}

// ---------------------------------------------------------------------------
// Kernel 1: fused QKV projection GEMM (2048x768 @ 768x2304), bf16 WMMA.
// Block: 256 thr (8 waves), 64x64 output tile. q,k -> bf16 padded [b,h,n,32];
// v -> f32 [b,n,768].
// ---------------------------------------------------------------------------
__global__ __launch_bounds__(256) void proj_kernel(
    const float* __restrict__ query,
    const float* __restrict__ Wq, const float* __restrict__ bq,
    const float* __restrict__ Wk, const float* __restrict__ bk,
    const float* __restrict__ Wv, const float* __restrict__ bv,
    unsigned short* __restrict__ q_bf, unsigned short* __restrict__ k_bf,
    float* __restrict__ v_f)
{
    __shared__ __bf16 As[64][34];   // activations tile, +2 pad vs bank conflicts
    __shared__ __bf16 Ws[64][34];   // weight tile: Ws[out_col][k]

    const int t = threadIdx.x;
    const int lane = t & 31, half = lane >> 4, r = lane & 15;
    const int w = t >> 5;
    const int mi = w & 3;          // wave row sub-tile (16 rows)
    const int ni = w >> 2;         // wave col sub-tile (32 cols)
    const int Mb = blockIdx.x * 64;          // global row base (b*512+n)
    const int Ob = blockIdx.y * 64;          // global out-col base in [0,2304)
    const int proj  = Ob / Ee;               // 0=q,1=k,2=v (64 | 768)
    const int ocol0 = Ob - proj * Ee;
    const float* __restrict__ W    = (proj == 0) ? Wq : (proj == 1) ? Wk : Wv;
    const float* __restrict__ bias = (proj == 0) ? bq : (proj == 1) ? bk : bv;

    v8f acc0 = {}; v8f acc1 = {};

    for (int kb = 0; kb < Ee; kb += 32) {
        for (int idx = t; idx < 64 * 32; idx += 256) {
            const int row = idx >> 5, kk = idx & 31;
            As[row][kk] = bits_to_bf16(f32_to_bf16_bits(query[(size_t)(Mb + row) * Ee + kb + kk]));
            Ws[row][kk] = bits_to_bf16(f32_to_bf16_bits(W[(size_t)(ocol0 + row) * Ee + kb + kk]));
        }
        __syncthreads();

        // A fragment: lane holds row (16*mi + r); K per ISA 7.12.2 layout.
        v16bf a;
        const int arow = 16 * mi + r;
#pragma unroll
        for (int j = 0; j < 8; ++j) {
            const int kk = (j < 4) ? (8 * half + 2 * j) : (16 + 8 * half + 2 * (j - 4));
            a[2 * j]     = As[arow][kk];
            a[2 * j + 1] = As[arow][kk + 1];
        }
        // B fragments: col = lane&15, K per ISA 7.12.2 layout (same as A).
        v16bf b0, b1;
        const int c0 = 32 * ni + r, c1 = 32 * ni + 16 + r;
#pragma unroll
        for (int j = 0; j < 8; ++j) {
            const int kk = (j < 4) ? (8 * half + 2 * j) : (16 + 8 * half + 2 * (j - 4));
            b0[2 * j] = Ws[c0][kk]; b0[2 * j + 1] = Ws[c0][kk + 1];
            b1[2 * j] = Ws[c1][kk]; b1[2 * j + 1] = Ws[c1][kk + 1];
        }
        acc0 = __builtin_amdgcn_wmma_f32_16x16x32_bf16(false, a, false, b0, (short)0, acc0, false, false);
        acc1 = __builtin_amdgcn_wmma_f32_16x16x32_bf16(false, a, false, b1, (short)0, acc1, false, false);
        __syncthreads();
    }

    const float scaling = rsqrtf((float)Dd);
#pragma unroll
    for (int tc = 0; tc < 2; ++tc) {
        v8f accv = tc ? acc1 : acc0;
#pragma unroll
        for (int i = 0; i < 8; ++i) {
            const int grow = Mb + 16 * mi + i + 8 * half;   // C layout: M = i + 8*half
            const int colL = 32 * ni + 16 * tc + r;         // N = lane&15
            const int o    = ocol0 + colL;
            const float val = accv[i] + bias[o];
            const int bb = grow >> 9, n = grow & 511;
            if (proj == 2) {
                v_f[(size_t)grow * Ee + o] = val;
            } else {
                const int hh = o / Dd, dk = o - hh * Dd;
                const size_t off = ((size_t)(bb * Hh + hh) * Nn + n) * DP + dk;
                if (proj == 0) q_bf[off] = f32_to_bf16_bits(val * scaling);
                else           k_bf[off] = f32_to_bf16_bits(val);
            }
        }
    }
}

// ---------------------------------------------------------------------------
// Kernel 2: w[b,h,m,c] = sum_d v[b,m,h*24+d] * Wf[c, h*24+d]   (tiny)
// ---------------------------------------------------------------------------
__global__ void wproj_kernel(const float* __restrict__ v_f,
                             const float* __restrict__ Wf,
                             float* __restrict__ w_f)
{
    const int idx = blockIdx.x * blockDim.x + threadIdx.x;   // (b*H + h)*N + m
    if (idx >= Bc * Hh * Nn) return;
    const int m = idx & (Nn - 1);
    const int h = (idx / Nn) & (Hh - 1);
    const int b = idx / (Nn * Hh);
    const float* vp = v_f + ((size_t)(b * Nn + m)) * Ee + h * Dd;
    const float* wf = Wf + h * Dd;
    float a0 = 0.f, a1 = 0.f, a2 = 0.f;
#pragma unroll
    for (int dk = 0; dk < Dd; ++dk) {
        const float vv = vp[dk];
        a0 += vv * wf[0 * Ee + dk];
        a1 += vv * wf[1 * Ee + dk];
        a2 += vv * wf[2 * Ee + dk];
    }
    float* wp = w_f + (size_t)idx * 3;
    wp[0] = a0; wp[1] = a1; wp[2] = a2;
}

// ---------------------------------------------------------------------------
// Kernel 3: flash-style attention + 3-axis contraction.
// One wave per (b, 16-row tile, head). Head's k slice (512x32 bf16, 32KB) is
// bulk-copied global->LDS with async-to-LDS B128 ops (ASYNCcnt), then each
// 16-key tile is 1 WMMA + online softmax + fused p*delta*w accumulation.
// ---------------------------------------------------------------------------
__global__ __launch_bounds__(32) void attn_kernel(
    const float* __restrict__ attn_bias, const float* __restrict__ delta_pos,
    const unsigned short* __restrict__ q_bf, const unsigned short* __restrict__ k_bf,
    const float* __restrict__ w_f, float* __restrict__ out)
{
    __shared__ __align__(16) unsigned short ksh[Nn * DP];   // 32 KB

    const int nt = blockIdx.x, h = blockIdx.y, b = blockIdx.z;
    const int lane = threadIdx.x & 31, half = lane >> 4, r = lane & 15;
    const int n0 = nt * 16;

    const size_t biasBase = (size_t)(b * Hh + h) * Nn * Nn;
    const size_t kBase    = (size_t)(b * Hh + h) * Nn * DP;
    const size_t wBase    = (size_t)(b * Hh + h) * Nn * 3;

    // ---- async bulk copy: k head slice (32 KB) global -> LDS --------------
    // INST_OFFSET==0; per-lane byte offset drives both LDS dest and global src
    // (GVS mode: saddr = uniform 64-bit base, vaddr = 32-bit byte offset).
    {
        const unsigned short* kg = k_bf + kBase;            // uniform SGPR base
        unsigned ldsbase = (unsigned)(uintptr_t)(void*)ksh; // low 32b = LDS offset
#pragma unroll 8
        for (int i = 0; i < 64; ++i) {                      // 64 x 32 lanes x 16B
            unsigned off  = (unsigned)(lane * 16 + i * 512);
            unsigned ldsa = ldsbase + off;
            asm volatile("global_load_async_to_lds_b128 %0, %1, %2 offset:0"
                         :: "v"(ldsa), "v"(off), "s"(kg) : "memory");
        }
    }

    // A fragment straight from global (padded layout makes it 2x 16B loads).
    union Frag { uint4 q[2]; v16bf v; };
    Frag au;
    {
        const uint4* qp = reinterpret_cast<const uint4*>(
            q_bf + ((size_t)(b * Hh + h) * Nn + (n0 + r)) * DP);
        au.q[0] = qp[half];        // bf idx 8*half .. 8*half+7   -> j=0..3
        au.q[1] = qp[2 + half];    // bf idx 16+8*half .. +7      -> j=4..7
    }

    float mx[8], sm[8], acc[8][3];
#pragma unroll
    for (int i = 0; i < 8; ++i) {
        mx[i] = -1e30f; sm[i] = 0.f;
        acc[i][0] = acc[i][1] = acc[i][2] = 0.f;
    }

    // single wave: ASYNCcnt==0 implies our LDS writes landed; no barrier needed
    asm volatile("s_wait_asynccnt 0x0" ::: "memory");

    for (int mt = 0; mt < 32; ++mt) {
        const int m = mt * 16 + r;
        // B fragment from LDS: col = r, K per ISA layout (same as A) -> 2x 16B ds reads.
        Frag bu;
        {
            const uint4* kls = reinterpret_cast<const uint4*>(&ksh[m * DP]);
            bu.q[0] = kls[half];
            bu.q[1] = kls[2 + half];
        }

        v8f cz = {};
        v8f s = __builtin_amdgcn_wmma_f32_16x16x32_bf16(false, au.v, false, bu.v,
                                                        (short)0, cz, false, false);
        const float w0 = w_f[wBase + (size_t)m * 3 + 0];
        const float w1 = w_f[wBase + (size_t)m * 3 + 1];
        const float w2 = w_f[wBase + (size_t)m * 3 + 2];

        if (mt + 1 < 32) {  // uniform: warm next bias row segment
            __builtin_prefetch(attn_bias + biasBase + (size_t)(n0 + 8 * half) * Nn + (mt + 1) * 16, 0, 1);
        }

#pragma unroll
        for (int i = 0; i < 8; ++i) {
            const int n = n0 + i + 8 * half;              // C layout row
            float sv = s[i] + attn_bias[biasBase + (size_t)n * Nn + m];
            float tm = sv;                                 // 16-lane row max
            tm = fmaxf(tm, __shfl_xor(tm, 1));
            tm = fmaxf(tm, __shfl_xor(tm, 2));
            tm = fmaxf(tm, __shfl_xor(tm, 4));
            tm = fmaxf(tm, __shfl_xor(tm, 8));
            const float nm = fmaxf(mx[i], tm);
            const float sc = __expf(mx[i] - nm);
            mx[i] = nm;
            const float p = __expf(sv - nm);
            sm[i] = sm[i] * sc + p;                        // lane-partial
            const float* dp = delta_pos + ((size_t)(b * Nn + n) * Nn + m) * 3;
            acc[i][0] = acc[i][0] * sc + p * dp[0] * w0;
            acc[i][1] = acc[i][1] * sc + p * dp[1] * w1;
            acc[i][2] = acc[i][2] * sc + p * dp[2] * w2;
        }
    }

#pragma unroll
    for (int i = 0; i < 8; ++i) {
        float S = sm[i];
        S += __shfl_xor(S, 1); S += __shfl_xor(S, 2);
        S += __shfl_xor(S, 4); S += __shfl_xor(S, 8);
        float a0 = acc[i][0], a1 = acc[i][1], a2 = acc[i][2];
        a0 += __shfl_xor(a0, 1); a0 += __shfl_xor(a0, 2); a0 += __shfl_xor(a0, 4); a0 += __shfl_xor(a0, 8);
        a1 += __shfl_xor(a1, 1); a1 += __shfl_xor(a1, 2); a1 += __shfl_xor(a1, 4); a1 += __shfl_xor(a1, 8);
        a2 += __shfl_xor(a2, 1); a2 += __shfl_xor(a2, 2); a2 += __shfl_xor(a2, 4); a2 += __shfl_xor(a2, 8);
        if (r == 0) {
            const int n = n0 + i + 8 * half;
            const float inv = 1.0f / S;
            float* op = out + (size_t)(b * Nn + n) * 3;
            atomicAdd(op + 0, a0 * inv);
            atomicAdd(op + 1, a1 * inv);
            atomicAdd(op + 2, a2 * inv);
        }
    }
}

// ---------------------------------------------------------------------------
extern "C" void kernel_launch(void* const* d_in, const int* in_sizes, int n_in,
                              void* d_out, int out_size, void* d_ws, size_t ws_size,
                              hipStream_t stream) {
    const float* query     = (const float*)d_in[0];
    const float* attn_bias = (const float*)d_in[1];
    const float* delta_pos = (const float*)d_in[2];
    const float* Wq = (const float*)d_in[3];  const float* bq = (const float*)d_in[4];
    const float* Wk = (const float*)d_in[5];  const float* bk = (const float*)d_in[6];
    const float* Wv = (const float*)d_in[7];  const float* bv = (const float*)d_in[8];
    const float* Wf = (const float*)d_in[9];  const float* bf = (const float*)d_in[10];
    float* out = (float*)d_out;

    // Workspace layout: q_bf [0,4Mi) | k_bf [4Mi,8Mi) | v_f32 [8Mi,14Mi) | w [14Mi,..)
    char* ws = (char*)d_ws;
    unsigned short* q_bf = (unsigned short*)(ws);
    unsigned short* k_bf = (unsigned short*)(ws + (4u << 20));
    float* v_f = (float*)(ws + (8u << 20));
    float* w_f = (float*)(ws + (14u << 20));

    // 0) zero padded q/k (8 MiB => 2M words), init out with bf broadcast
    init_kernel<<<1024, 256, 0, stream>>>((unsigned*)ws, (size_t)(8u << 20) / 4, out, bf);

    // 1) fused QKV projection: M=2048 rows, 2304 out cols, 64x64 tiles
    proj_kernel<<<dim3(2048 / 64, 2304 / 64), 256, 0, stream>>>(
        query, Wq, bq, Wk, bk, Wv, bv, q_bf, k_bf, v_f);

    // 2) per-(b,h,m) 3-axis value weights
    wproj_kernel<<<(Bc * Hh * Nn + 255) / 256, 256, 0, stream>>>(v_f, Wf, w_f);

    // 3) flash attention + contraction, one wave per (row-tile, head, batch)
    attn_kernel<<<dim3(Nn / 16, Hh, Bc), 32, 0, stream>>>(
        attn_bias, delta_pos, q_bf, k_bf, w_f, out);
}
